// HEPN_38448547234283
// MI455X (gfx1250) — matrix operand order repaired and checked by reading, not compiled, not run on hardware
//
#include <hip/hip_runtime.h>

#define Nn 50000
#define Mm 800000
#define NLAYERS 2

typedef __attribute__((ext_vector_type(16))) _Float16 v16h;
typedef __attribute__((ext_vector_type(8)))  float    v8f;

__device__ __forceinline__ v8f vzero8() {
    v8f z = {0.f, 0.f, 0.f, 0.f, 0.f, 0.f, 0.f, 0.f};
    return z;
}

// Load a 16-half WMMA fragment from LDS as two 16B chunks (base, base+off2 halves).
// A- and B-fragments both use off2=16 (halves 8..15 hold K+16..K+23).
__device__ __forceinline__ v16h ldfrag(const _Float16* p, int off2) {
    union { uint4 q[2]; v16h v; } u;
    u.q[0] = *reinterpret_cast<const uint4*>(p);
    u.q[1] = *reinterpret_cast<const uint4*>(p + off2);
    return u.v;
}

__device__ __forceinline__ float silu_f(float v) {
    return v / (1.0f + __expf(-v));
}

// ---------------------------------------------------------------------------
// zero scratch accumulators (tot_msg | sum_f | cnt laid out contiguously)
// ---------------------------------------------------------------------------
__global__ __launch_bounds__(256) void zero_kernel(float* __restrict__ p, int n) {
    int i = blockIdx.x * 256 + threadIdx.x;
    if (i < n) p[i] = 0.0f;
}

// ---------------------------------------------------------------------------
// h = h_in @ emb_W + emb_b ;  also produce f16 copy for WMMA gathers
// ---------------------------------------------------------------------------
__global__ __launch_bounds__(256) void embed_kernel(
    const float* __restrict__ hin, const float* __restrict__ W,
    const float* __restrict__ b, float* __restrict__ hout,
    _Float16* __restrict__ h16, int n)
{
    __shared__ float sW[16 * 64];
    __shared__ float sB[64];
    int tid = threadIdx.x;
    for (int i = tid; i < 16 * 64; i += 256) sW[i] = W[i];
    if (tid < 64) sB[tid] = b[tid];
    __syncthreads();
    int nd = blockIdx.x * 256 + tid;
    if (nd >= n) return;
    float hr[16];
#pragma unroll
    for (int k = 0; k < 16; k++) hr[k] = hin[nd * 16 + k];
#pragma unroll 4
    for (int o = 0; o < 64; o++) {
        float a = sB[o];
#pragma unroll
        for (int k = 0; k < 16; k++) a += hr[k] * sW[k * 64 + o];
        hout[(size_t)nd * 64 + o] = a;
        h16[(size_t)nd * 64 + o] = (_Float16)a;
    }
}

// ---------------------------------------------------------------------------
// Edge kernel: persistent blocks, grid-stride over 64-edge groups; per wave,
// one 16-edge tile per iteration.
//  s = [h16[row](64) | h16[col](64) | edge_fea(8) | r2(1) | 0-pad] (K=160)
//  msg = SiLU(SiLU(s@W1+b1)@W2+b2)            -> atomicAdd tot_msg[row]
//  cm  = SiLU(msg@cW1+cb1)@cW2+cb2            -> atomicAdd sum_f[row] += rij*cm
// Weights staged transposed in LDS (Wt[n][k]) ONCE per block.
// ---------------------------------------------------------------------------
__global__ __launch_bounds__(128) void edge_kernel(
    const float* __restrict__ x, const _Float16* __restrict__ h16,
    const int* __restrict__ row, const int* __restrict__ col,
    const float* __restrict__ efea,
    const float* __restrict__ W1, const float* __restrict__ b1,
    const float* __restrict__ W2, const float* __restrict__ b2,
    const float* __restrict__ cW1, const float* __restrict__ cb1,
    const float* __restrict__ cW2, const float* __restrict__ cb2,
    float* __restrict__ tot_msg, float* __restrict__ sum_f,
    float* __restrict__ cnt, int M)
{
    __shared__ __align__(16) _Float16 sW1[64 * 168];   // W1t, K padded to 160, pitch 168
    __shared__ __align__(16) _Float16 sW2[64 * 72];    // W2t, pitch 72
    __shared__ __align__(16) _Float16 sC1[64 * 72];    // coord_W1t
    __shared__ float sB1[64], sB2[64], sCB1[64], sCW2v[64];
    __shared__ float sCB2;
    __shared__ __align__(16) _Float16 sA[4][16 * 168]; // per-wave s tile
    __shared__ __align__(16) _Float16 sT0[4][16 * 72]; // per-wave intermediate tile
    __shared__ int   sRi[4][16];
    __shared__ float sRij[4][48];

    int tid = threadIdx.x;
    // stage W1 transposed + column-permuted: new k<136 -> old row k+1, k==136 -> old row 0 (r2)
    for (int i = tid; i < 64 * 168; i += 128) {
        int n = i / 168, k = i - n * 168;
        float w = 0.0f;
        if (k < 136)       w = W1[(k + 1) * 64 + n];
        else if (k == 136) w = W1[n];
        sW1[i] = (_Float16)w;
    }
    for (int i = tid; i < 64 * 72; i += 128) {
        int n = i / 72, k = i - n * 72;
        sW2[i] = (_Float16)(k < 64 ? W2[k * 64 + n] : 0.0f);
        sC1[i] = (_Float16)(k < 64 ? cW1[k * 64 + n] : 0.0f);
    }
    if (tid < 64) { sB1[tid] = b1[tid]; sB2[tid] = b2[tid]; sCB1[tid] = cb1[tid]; sCW2v[tid] = cW2[tid]; }
    if (tid == 0) sCB2 = cb2[0];

    int wid = tid >> 5, lane = tid & 31;
    int ln = lane & 15, hi = lane >> 4;
    int ngroups = (M + 63) >> 6;

    for (int g = blockIdx.x; g < ngroups; g += gridDim.x) {
        int e0 = g * 64 + wid * 16;

        // geometry + tail columns (one lane per edge)
        if (lane < 16) {
            int e = e0 + lane;
            bool v = e < M;
            int r = 0, c = 0;
            if (v) { r = row[e]; c = col[e]; }
            sRi[wid][lane] = r;
            float dx = x[r * 3 + 0] - x[c * 3 + 0];
            float dy = x[r * 3 + 1] - x[c * 3 + 1];
            float dz = x[r * 3 + 2] - x[c * 3 + 2];
            if (!v) { dx = dy = dz = 0.0f; }
            sRij[wid][lane * 3 + 0] = dx;
            sRij[wid][lane * 3 + 1] = dy;
            sRij[wid][lane * 3 + 2] = dz;
            float r2 = dx * dx + dy * dy + dz * dz;
            _Float16* srow = &sA[wid][lane * 168];
#pragma unroll
            for (int j = 0; j < 8; j++) srow[128 + j] = (_Float16)(v ? efea[(size_t)e * 8 + j] : 0.0f);
            srow[136] = (_Float16)r2;
            for (int k = 137; k < 168; k++) srow[k] = (_Float16)0.0f;
        }
        // gather h16[row]/h16[col] -> cols 0..127 (2 lanes per edge, 8x b128 each)
        {
            int e = lane >> 1, sel = lane & 1;
            int ge = e0 + e;
            int idx = 0;
            if (ge < M) idx = sel ? col[ge] : row[ge];
            const uint4* src = reinterpret_cast<const uint4*>(h16 + (size_t)idx * 64);
            uint4* dst = reinterpret_cast<uint4*>(&sA[wid][e * 168 + sel * 64]);
#pragma unroll
            for (int j = 0; j < 8; j++) dst[j] = src[j];
        }
        __syncthreads();

        v8f acc[4];
        // ---- GEMM1: 16x160 @ 160x64 -> 16x64 (20 WMMA) ----
#pragma unroll
        for (int t = 0; t < 4; t++) acc[t] = vzero8();
        {
            const _Float16* Ab = &sA[wid][ln * 168 + hi * 8];
#pragma unroll
            for (int kt = 0; kt < 5; kt++) {
                v16h a = ldfrag(Ab + kt * 32, 16);
#pragma unroll
                for (int nt = 0; nt < 4; nt++) {
                    v16h bf = ldfrag(&sW1[(nt * 16 + ln) * 168 + kt * 32 + hi * 8], 16);
                    acc[nt] = __builtin_amdgcn_wmma_f32_16x16x32_f16(
                        false, a, false, bf, (short)0, acc[nt], false, false);
                }
            }
        }
#pragma unroll
        for (int nt = 0; nt < 4; nt++) {
            float bia = sB1[nt * 16 + ln];
#pragma unroll
            for (int v = 0; v < 8; v++)
                sT0[wid][(hi * 8 + v) * 72 + nt * 16 + ln] = (_Float16)silu_f(acc[nt][v] + bia);
        }
        __syncthreads();

        // ---- GEMM2: 16x64 @ 64x64 -> msg (8 WMMA) ----
#pragma unroll
        for (int t = 0; t < 4; t++) acc[t] = vzero8();
        {
            const _Float16* Ab = &sT0[wid][ln * 72 + hi * 8];
#pragma unroll
            for (int kt = 0; kt < 2; kt++) {
                v16h a = ldfrag(Ab + kt * 32, 16);
#pragma unroll
                for (int nt = 0; nt < 4; nt++) {
                    v16h bf = ldfrag(&sW2[(nt * 16 + ln) * 72 + kt * 32 + hi * 8], 16);
                    acc[nt] = __builtin_amdgcn_wmma_f32_16x16x32_f16(
                        false, a, false, bf, (short)0, acc[nt], false, false);
                }
            }
        }
        // msg = SiLU(acc+b2): scatter to tot_msg from registers, restage into sT0 in place
#pragma unroll
        for (int nt = 0; nt < 4; nt++) {
            float bia = sB2[nt * 16 + ln];
#pragma unroll
            for (int v = 0; v < 8; v++) {
                int Mr = hi * 8 + v;
                float m = silu_f(acc[nt][v] + bia);
                sT0[wid][Mr * 72 + nt * 16 + ln] = (_Float16)m;
                int e = e0 + Mr;
                if (e < M)
                    atomicAdd(&tot_msg[(size_t)sRi[wid][Mr] * 64 + nt * 16 + ln], m);
            }
        }
        __syncthreads();

        // ---- coord hidden: 16x64 @ 64x64 (8 WMMA), then dot with cW2 ----
#pragma unroll
        for (int t = 0; t < 4; t++) acc[t] = vzero8();
        {
            const _Float16* Ab = &sT0[wid][ln * 72 + hi * 8];
#pragma unroll
            for (int kt = 0; kt < 2; kt++) {
                v16h a = ldfrag(Ab + kt * 32, 16);
#pragma unroll
                for (int nt = 0; nt < 4; nt++) {
                    v16h bf = ldfrag(&sC1[(nt * 16 + ln) * 72 + kt * 32 + hi * 8], 16);
                    acc[nt] = __builtin_amdgcn_wmma_f32_16x16x32_f16(
                        false, a, false, bf, (short)0, acc[nt], false, false);
                }
            }
        }
        // cm[Mr] = sum_n SiLU(hid)*cW2[n] + cb2 ; butterfly-reduce over 16 lanes/half
        float cm[8];
#pragma unroll
        for (int v = 0; v < 8; v++) {
            float p = 0.0f;
#pragma unroll
            for (int nt = 0; nt < 4; nt++) {
                int n = nt * 16 + ln;
                p += silu_f(acc[nt][v] + sCB1[n]) * sCW2v[n];
            }
#pragma unroll
            for (int m = 8; m >= 1; m >>= 1) p += __shfl_xor(p, m, 32);
            cm[v] = p + sCB2;
        }
        if (ln == 0) {
#pragma unroll
            for (int v = 0; v < 8; v++) {
                int Mr = hi * 8 + v;
                int e = e0 + Mr;
                if (e < M) {
                    int r = sRi[wid][Mr];
                    float c = cm[v];
                    atomicAdd(&sum_f[r * 3 + 0], sRij[wid][Mr * 3 + 0] * c);
                    atomicAdd(&sum_f[r * 3 + 1], sRij[wid][Mr * 3 + 1] * c);
                    atomicAdd(&sum_f[r * 3 + 2], sRij[wid][Mr * 3 + 2] * c);
                    atomicAdd(&cnt[r], 1.0f);
                }
            }
        }
        __syncthreads();
    }
}

// ---------------------------------------------------------------------------
// Node kernel: persistent blocks, grid-stride over 64-node groups.
// x += clip(sum_f / max(cnt,1)); h = [h16|tot_msg] @ W1 silu @ W2 + b2
// ---------------------------------------------------------------------------
__global__ __launch_bounds__(128) void node_kernel(
    float* __restrict__ x, const float* __restrict__ sum_f, const float* __restrict__ cnt,
    _Float16* __restrict__ h16, const float* __restrict__ tot_msg,
    const float* __restrict__ W1, const float* __restrict__ b1,
    const float* __restrict__ W2, const float* __restrict__ b2,
    float* __restrict__ hout, int n)
{
    __shared__ __align__(16) _Float16 sW1[64 * 136]; // W1t, K=128 padded, pitch 136
    __shared__ __align__(16) _Float16 sW2[64 * 72];
    __shared__ float sB1[64], sB2[64];
    __shared__ __align__(16) _Float16 sA[4][16 * 136];
    __shared__ __align__(16) _Float16 sT0[4][16 * 72];

    int tid = threadIdx.x;
    for (int i = tid; i < 64 * 136; i += 128) {
        int nn = i / 136, k = i - nn * 136;
        sW1[i] = (_Float16)(k < 128 ? W1[k * 64 + nn] : 0.0f);
    }
    for (int i = tid; i < 64 * 72; i += 128) {
        int nn = i / 72, k = i - nn * 72;
        sW2[i] = (_Float16)(k < 64 ? W2[k * 64 + nn] : 0.0f);
    }
    if (tid < 64) { sB1[tid] = b1[tid]; sB2[tid] = b2[tid]; }

    int wid = tid >> 5, lane = tid & 31;
    int ln = lane & 15, hi = lane >> 4;
    int ngroups = (n + 63) >> 6;

    for (int g = blockIdx.x; g < ngroups; g += gridDim.x) {
        int n0 = g * 64 + wid * 16;

        // coordinate update (mean aggregation + clip)
        if (lane < 16) {
            int nd = n0 + lane;
            if (nd < n) {
                float c = fmaxf(cnt[nd], 1.0f);
#pragma unroll
                for (int d = 0; d < 3; d++) {
                    float t = sum_f[nd * 3 + d] / c;
                    t = fminf(fmaxf(t, -100.0f), 100.0f);
                    x[nd * 3 + d] += t;
                }
            }
        }
        // stage s2 = [h16 | tot_msg] (2 lanes per node)
        {
            int e = lane >> 1, sel = lane & 1;
            int nd = n0 + e;
            if (nd >= n) nd = 0;
            if (sel == 0) {
                const uint4* src = reinterpret_cast<const uint4*>(h16 + (size_t)nd * 64);
                uint4* dst = reinterpret_cast<uint4*>(&sA[wid][e * 136]);
#pragma unroll
                for (int j = 0; j < 8; j++) dst[j] = src[j];
            } else {
                const float* src = tot_msg + (size_t)nd * 64;
                _Float16* dst = &sA[wid][e * 136 + 64];
#pragma unroll 8
                for (int j = 0; j < 64; j++) dst[j] = (_Float16)src[j];
#pragma unroll
                for (int j = 0; j < 8; j++) dst[64 + j] = (_Float16)0.0f;
            }
        }
        __syncthreads();

        v8f acc[4];
        // GEMM1: 16x128 @ 128x64 (16 WMMA)
#pragma unroll
        for (int t = 0; t < 4; t++) acc[t] = vzero8();
        {
            const _Float16* Ab = &sA[wid][ln * 136 + hi * 8];
#pragma unroll
            for (int kt = 0; kt < 4; kt++) {
                v16h a = ldfrag(Ab + kt * 32, 16);
#pragma unroll
                for (int nt = 0; nt < 4; nt++) {
                    v16h bf = ldfrag(&sW1[(nt * 16 + ln) * 136 + kt * 32 + hi * 8], 16);
                    acc[nt] = __builtin_amdgcn_wmma_f32_16x16x32_f16(
                        false, a, false, bf, (short)0, acc[nt], false, false);
                }
            }
        }
#pragma unroll
        for (int nt = 0; nt < 4; nt++) {
            float bia = sB1[nt * 16 + ln];
#pragma unroll
            for (int v = 0; v < 8; v++)
                sT0[wid][(hi * 8 + v) * 72 + nt * 16 + ln] = (_Float16)silu_f(acc[nt][v] + bia);
        }
        __syncthreads();

        // GEMM2: 16x64 @ 64x64, no final activation (8 WMMA)
#pragma unroll
        for (int t = 0; t < 4; t++) acc[t] = vzero8();
        {
            const _Float16* Ab = &sT0[wid][ln * 72 + hi * 8];
#pragma unroll
            for (int kt = 0; kt < 2; kt++) {
                v16h a = ldfrag(Ab + kt * 32, 16);
#pragma unroll
                for (int nt = 0; nt < 4; nt++) {
                    v16h bf = ldfrag(&sW2[(nt * 16 + ln) * 72 + kt * 32 + hi * 8], 16);
                    acc[nt] = __builtin_amdgcn_wmma_f32_16x16x32_f16(
                        false, a, false, bf, (short)0, acc[nt], false, false);
                }
            }
        }
#pragma unroll
        for (int nt = 0; nt < 4; nt++) {
            float bia = sB2[nt * 16 + ln];
#pragma unroll
            for (int v = 0; v < 8; v++) {
                int Mr = hi * 8 + v;
                int nd = n0 + Mr;
                if (nd < n) {
                    float val = acc[nt][v] + bia;
                    hout[(size_t)nd * 64 + nt * 16 + ln] = val;
                    h16[(size_t)nd * 64 + nt * 16 + ln] = (_Float16)val;
                }
            }
        }
        __syncthreads();
    }
}

// ---------------------------------------------------------------------------
extern "C" void kernel_launch(void* const* d_in, const int* in_sizes, int n_in,
                              void* d_out, int out_size, void* d_ws, size_t ws_size,
                              hipStream_t stream) {
    (void)in_sizes; (void)n_in; (void)out_size; (void)ws_size;
    const float* x_in = (const float*)d_in[0];
    const float* h_in = (const float*)d_in[1];
    const int*   row  = (const int*)d_in[2];
    const int*   col  = (const int*)d_in[3];
    const float* efea = (const float*)d_in[4];
    const float* embW = (const float*)d_in[5];
    const float* embB = (const float*)d_in[6];
    const float* eW1  = (const float*)d_in[7];
    const float* eb1  = (const float*)d_in[8];
    const float* eW2  = (const float*)d_in[9];
    const float* eb2  = (const float*)d_in[10];
    const float* cW1  = (const float*)d_in[11];
    const float* cb1  = (const float*)d_in[12];
    const float* cW2  = (const float*)d_in[13];
    const float* cb2  = (const float*)d_in[14];
    const float* nW1  = (const float*)d_in[15];
    const float* nb1  = (const float*)d_in[16];
    const float* nW2  = (const float*)d_in[17];
    const float* nb2  = (const float*)d_in[18];

    // workspace layout (floats): x_cur | h_f | tot_msg | sum_f | cnt | h16(half)
    float* x_cur   = (float*)d_ws;
    float* h_f     = x_cur + (size_t)Nn * 3;
    float* tot_msg = h_f + (size_t)Nn * 64;     // tot_msg|sum_f|cnt contiguous -> one zero pass
    float* sum_f   = tot_msg + (size_t)Nn * 64;
    float* cnt     = sum_f + (size_t)Nn * 3;
    _Float16* h16  = (_Float16*)(cnt + Nn);

    hipMemcpyAsync(x_cur, x_in, (size_t)Nn * 3 * sizeof(float),
                   hipMemcpyDeviceToDevice, stream);
    embed_kernel<<<(Nn + 255) / 256, 256, 0, stream>>>(h_in, embW, embB, h_f, h16, Nn);

    // persistent-block grids: amortize per-block LDS weight staging
    const int egroups = (Mm + 63) / 64;
    const int ngroups = (Nn + 63) / 64;
    const int eblocks = egroups < 1024 ? egroups : 1024; // ~12 groups per block
    const int nblocks = ngroups < 256 ? ngroups : 256;   // ~3 groups per block
    const int zcount  = Nn * 68;                         // tot_msg(64)+sum_f(3)+cnt(1)

    for (int l = 0; l < NLAYERS; l++) {
        zero_kernel<<<(zcount + 255) / 256, 256, 0, stream>>>(tot_msg, zcount);
        edge_kernel<<<eblocks, 128, 0, stream>>>(
            x_cur, h16, row, col, efea,
            eW1 + (size_t)l * 137 * 64, eb1 + l * 64,
            eW2 + (size_t)l * 64 * 64,  eb2 + l * 64,
            cW1 + (size_t)l * 64 * 64,  cb1 + l * 64,
            cW2 + l * 64,               cb2 + l,
            tot_msg, sum_f, cnt, Mm);
        node_kernel<<<nblocks, 128, 0, stream>>>(
            x_cur, sum_f, cnt, h16, tot_msg,
            nW1 + (size_t)l * 128 * 64, nb1 + l * 64,
            nW2 + (size_t)l * 64 * 64,  nb2 + l * 64,
            h_f, Nn);
    }

    hipMemcpyAsync(d_out, x_cur, (size_t)Nn * 3 * sizeof(float),
                   hipMemcpyDeviceToDevice, stream);
    hipMemcpyAsync((float*)d_out + (size_t)Nn * 3, h_f,
                   (size_t)Nn * 64 * sizeof(float), hipMemcpyDeviceToDevice, stream);
}
